// UKF_52561809769213
// MI455X (gfx1250) — matrix operand order repaired and checked by reading, not compiled, not run on hardware
//
#include <hip/hip_runtime.h>

typedef _Float16 v16h __attribute__((ext_vector_type(16)));
typedef _Float16 v8h  __attribute__((ext_vector_type(8)));
typedef float    v8f  __attribute__((ext_vector_type(8)));

#define TPB 256

// ---------------- CDNA5 async global->LDS copy (ASYNCcnt path) ----------------
__device__ __forceinline__ void async_b128(uint32_t lds_byte_addr, const _Float16* g) {
  // VDST = per-lane LDS byte address, VADDR = 64-bit global address, no SADDR
  asm volatile("global_load_async_to_lds_b128 %0, %1, off"
               :: "v"(lds_byte_addr), "v"((uint64_t)(uintptr_t)g)
               : "memory");
}
__device__ __forceinline__ void wait_async0() {
  asm volatile("s_wait_asynccnt 0" ::: "memory");
}

__device__ __forceinline__ v16h pack16(v8h lo, v8h hi) {
  v16h r;
#pragma unroll
  for (int t = 0; t < 8; ++t) { r[t] = lo[t]; r[t + 8] = hi[t]; }
  return r;
}

// ---------------- generic deterministic reductions ----------------
__global__ __launch_bounds__(TPB) void k_sum_partial(const float* __restrict__ in, size_t n,
                                                     float* __restrict__ part) {
  __shared__ float red[TPB];
  float acc = 0.f;
  for (size_t i = (size_t)blockIdx.x * TPB + threadIdx.x; i < n; i += (size_t)gridDim.x * TPB)
    acc += in[i];
  red[threadIdx.x] = acc; __syncthreads();
  for (int s = TPB / 2; s > 0; s >>= 1) {
    if ((int)threadIdx.x < s) red[threadIdx.x] += red[threadIdx.x + s];
    __syncthreads();
  }
  if (threadIdx.x == 0) part[blockIdx.x] = red[0];
}

__global__ __launch_bounds__(TPB) void k_freduce(const float* __restrict__ in, int n,
                                                 float* __restrict__ out) {
  __shared__ float red[TPB];
  float acc = 0.f;
  for (int i = threadIdx.x; i < n; i += TPB) acc += in[i];
  red[threadIdx.x] = acc; __syncthreads();
  for (int s = TPB / 2; s > 0; s >>= 1) {
    if ((int)threadIdx.x < s) red[threadIdx.x] += red[threadIdx.x + s];
    __syncthreads();
  }
  if (threadIdx.x == 0) *out = red[0];
}

// ---------------- scalar glue ----------------
__global__ void k_scalar_s1(const float* sumP, float w, float* s1) { *s1 = 2.f * w * w * (*sumP); }
__global__ void k_scalar_s23(const float* a2, const float* a3, float w, float* s2, float* s3) {
  *s2 = 2.f * w * (*a2);
  *s3 = 2.f * w * (*a3);
}

// ---------------- matvec: out = M1*v1 + M2*v2 + c ----------------
__global__ __launch_bounds__(TPB) void k_matvec2(const float* __restrict__ M1, const float* __restrict__ v1,
                                                 const float* __restrict__ M2, const float* __restrict__ v2,
                                                 const float* __restrict__ c, float* __restrict__ out, int d) {
  __shared__ float red[TPB];
  const int row = blockIdx.x;
  const float* r1 = M1 + (size_t)row * d;
  const float* r2 = M2 + (size_t)row * d;
  float acc = 0.f;
  for (int j = threadIdx.x; j < d; j += TPB) acc += r1[j] * v1[j] + r2[j] * v2[j];
  red[threadIdx.x] = acc; __syncthreads();
  for (int s = TPB / 2; s > 0; s >>= 1) {
    if ((int)threadIdx.x < s) red[threadIdx.x] += red[threadIdx.x + s];
    __syncthreads();
  }
  if (threadIdx.x == 0) out[row] = red[0] + c[row];
}

// ---------------- builders / converters ----------------
__global__ __launch_bounds__(TPB) void k_build_s2h(const float* __restrict__ Q, const float* __restrict__ s1p,
                                                   float w, _Float16* __restrict__ S2h, int d) {
  const size_t idx = (size_t)blockIdx.x * TPB + threadIdx.x;
  const int i = (int)(idx / (size_t)d), j = (int)(idx % (size_t)d);
  S2h[idx] = (_Float16)sqrtf(w * (Q[(size_t)j * d + i] + *s1p));
}

__global__ __launch_bounds__(TPB) void k_f2h(const float* __restrict__ in, _Float16* __restrict__ out, size_t n) {
  const size_t idx = (size_t)blockIdx.x * TPB + threadIdx.x;
  if (idx < n) out[idx] = (_Float16)in[idx];
}

__global__ __launch_bounds__(TPB) void k_build_pyh(const float* __restrict__ R, const float* __restrict__ s2p,
                                                   _Float16* __restrict__ PyH, size_t n) {
  const size_t idx = (size_t)blockIdx.x * TPB + threadIdx.x;
  if (idx < n) PyH[idx] = (_Float16)(R[idx] + *s2p);
}

__global__ __launch_bounds__(TPB) void k_init_X(const float* __restrict__ R, const float* __restrict__ s2p,
                                                const float* __restrict__ ap, float* __restrict__ X, int d) {
  const size_t idx = (size_t)blockIdx.x * TPB + threadIdx.x;
  const int i = (int)(idx / (size_t)d), j = (int)(idx % (size_t)d);
  X[idx] = (*ap) * (R[(size_t)j * d + i] + *s2p);
}

__global__ __launch_bounds__(TPB) void k_conv_XT(const float* __restrict__ X, _Float16* __restrict__ Xh,
                                                 _Float16* __restrict__ XTh, int d) {
  const size_t idx = (size_t)blockIdx.x * TPB + threadIdx.x;
  const int i = (int)(idx / (size_t)d), j = (int)(idx % (size_t)d);
  const _Float16 h = (_Float16)X[idx];
  Xh[idx] = h;
  XTh[(size_t)j * d + i] = h;
}

__global__ __launch_bounds__(TPB) void k_convT(const float* __restrict__ Z, _Float16* __restrict__ ZTh, int d) {
  const size_t idx = (size_t)blockIdx.x * TPB + threadIdx.x;
  const int i = (int)(idx / (size_t)d), j = (int)(idx % (size_t)d);
  ZTh[(size_t)j * d + i] = (_Float16)Z[idx];
}

// ---------------- row/col sums for ||Py||_inf, ||Py||_1 (entries positive) ----------------
__global__ __launch_bounds__(TPB) void k_rowsum(const float* __restrict__ R, float* __restrict__ rs, int d) {
  __shared__ float red[TPB];
  const float* row = R + (size_t)blockIdx.x * d;
  float acc = 0.f;
  for (int j = threadIdx.x; j < d; j += TPB) acc += row[j];
  red[threadIdx.x] = acc; __syncthreads();
  for (int s = TPB / 2; s > 0; s >>= 1) {
    if ((int)threadIdx.x < s) red[threadIdx.x] += red[threadIdx.x + s];
    __syncthreads();
  }
  if (threadIdx.x == 0) rs[blockIdx.x] = red[0];
}

__global__ __launch_bounds__(TPB) void k_colsum(const float* __restrict__ R, float* __restrict__ cs, int d) {
  __shared__ float red[TPB];
  float acc = 0.f;
  for (int i = threadIdx.x; i < d; i += TPB) acc += R[(size_t)i * d + blockIdx.x];
  red[threadIdx.x] = acc; __syncthreads();
  for (int s = TPB / 2; s > 0; s >>= 1) {
    if ((int)threadIdx.x < s) red[threadIdx.x] += red[threadIdx.x + s];
    __syncthreads();
  }
  if (threadIdx.x == 0) cs[blockIdx.x] = red[0];
}

__global__ __launch_bounds__(TPB) void k_alpha(const float* __restrict__ rs, const float* __restrict__ cs,
                                               const float* __restrict__ s2p, int d, float* __restrict__ alpha) {
  __shared__ float m1[TPB], m2[TPB];
  float a = -1e30f, b = -1e30f;
  for (int i = threadIdx.x; i < d; i += TPB) { a = fmaxf(a, rs[i]); b = fmaxf(b, cs[i]); }
  m1[threadIdx.x] = a; m2[threadIdx.x] = b; __syncthreads();
  for (int s = TPB / 2; s > 0; s >>= 1) {
    if ((int)threadIdx.x < s) {
      m1[threadIdx.x] = fmaxf(m1[threadIdx.x], m1[threadIdx.x + s]);
      m2[threadIdx.x] = fmaxf(m2[threadIdx.x], m2[threadIdx.x + s]);
    }
    __syncthreads();
  }
  if (threadIdx.x == 0) {
    const float s2 = *s2p;
    *alpha = 1.f / ((m1[0] + d * s2) * (m2[0] + d * s2));
  }
}

// =====================================================================
// LDS-staged WMMA NT-GEMM:  C[i][j] = sum_k A[i][k]*B[j][k]  (A,B f16 row-major)
// Block: 8 waves (2x4), wave tile 32x32 (2x2 WMMA tiles), block tile 64x128, BK=32.
// Global->LDS via async b128 copies (ASYNCcnt), double-buffered.
// MODE 0: no C output; per-wave partials of sum(acc^2), sum(S2h .* acc)
// MODE 1: store acc to f32 Out
// MODE 2: Out = 2*Xold - acc  (Newton-Schulz update)
// =====================================================================
#define AROW_H 40            // padded row stride in halves (32 data + 8 pad)
#define ABUF_H (64 * AROW_H)     // 2560 halves per A buffer
#define BBUF_H (128 * AROW_H)    // 5120 halves per B buffer
#define AROW_B 80            // bytes
#define ABUF_B 5120          // bytes
#define BBUF_B 10240         // bytes

template <int MODE>
__global__ __launch_bounds__(TPB) void k_gemm_tiled(const _Float16* __restrict__ Am,
                                                    const _Float16* __restrict__ Bm, int d,
                                                    const _Float16* __restrict__ Sh,
                                                    float* __restrict__ partA, float* __restrict__ partB,
                                                    const float* __restrict__ Xold, float* __restrict__ Out) {
  __shared__ __align__(16) _Float16 smem[2 * ABUF_H + 2 * BBUF_H];  // 30 KB

  const int tid  = threadIdx.x;
  const int lane = tid & 31;
  const int wid  = tid >> 5;
  const int m    = lane & 15;
  const int sel  = lane >> 4;
  const int wr   = wid >> 2;   // 0..1  -> row offset wr*32
  const int wc   = wid & 3;    // 0..3  -> col offset wc*32

  const int nbj = d / 128;
  const int bi  = blockIdx.x / nbj;
  const int bj  = blockIdx.x % nbj;

  const uint32_t lds0 = (uint32_t)(uintptr_t)smem;     // LDS byte base
  const int lrow = tid >> 2;                            // 0..63
  const int seg  = tid & 3;                             // 16B segment

  const _Float16* gA  = Am + (size_t)(bi * 64 + lrow) * d + seg * 8;
  const _Float16* gB0 = Bm + (size_t)(bj * 128 + lrow) * d + seg * 8;
  const _Float16* gB1 = Bm + (size_t)(bj * 128 + 64 + lrow) * d + seg * 8;
  const uint32_t lA  = lds0 + (uint32_t)(lrow * AROW_B + seg * 16);
  const uint32_t lB0 = lds0 + 2 * ABUF_B + (uint32_t)(lrow * AROW_B + seg * 16);
  const uint32_t lB1 = lB0 + (uint32_t)(64 * AROW_B);

  v8f acc[2][2] = {};

  // prologue: fill buffer 0
  async_b128(lA, gA);
  async_b128(lB0, gB0);
  async_b128(lB1, gB1);

  const int nsteps = d / 32;
  for (int s = 0; s < nsteps; ++s) {
    wait_async0();
    __syncthreads();
    if (s + 1 < nsteps) {
      const int nb = (s + 1) & 1;
      const int k0 = (s + 1) * 32;
      async_b128(lA + nb * ABUF_B, gA + k0);
      async_b128(lB0 + nb * BBUF_B, gB0 + k0);
      async_b128(lB1 + nb * BBUF_B, gB1 + k0);
    }
    const int buf = s & 1;
    const _Float16* As = smem + buf * ABUF_H;
    const _Float16* Bs = smem + 2 * ABUF_H + buf * BBUF_H;

    v16h af[2], bf[2];
#pragma unroll
    for (int t = 0; t < 2; ++t) {
      const _Float16* ar = As + (wr * 32 + t * 16 + m) * AROW_H;
      af[t] = pack16(*(const v8h*)(ar + sel * 8), *(const v8h*)(ar + 16 + sel * 8));
      const _Float16* br = Bs + (wc * 32 + t * 16 + m) * AROW_H;
      bf[t] = pack16(*(const v8h*)(br + sel * 8), *(const v8h*)(br + 16 + sel * 8));
    }
#pragma unroll
    for (int tm = 0; tm < 2; ++tm)
#pragma unroll
      for (int tn = 0; tn < 2; ++tn)
        acc[tm][tn] = __builtin_amdgcn_wmma_f32_16x16x32_f16(false, af[tm], false, bf[tn],
                                                             (short)0, acc[tm][tn], false, false);
  }

  // C/D layout: VGPR r, lanes 0-15 -> M=r, lanes 16-31 -> M=8+r; N = lane&15
  if (MODE == 0) {
    float s2p = 0.f, s3p = 0.f;
#pragma unroll
    for (int tm = 0; tm < 2; ++tm) {
#pragma unroll
      for (int tn = 0; tn < 2; ++tn) {
        const int grow = bi * 64 + wr * 32 + tm * 16 + sel * 8;
        const int gcol = bj * 128 + wc * 32 + tn * 16 + m;
        const _Float16* Sb = Sh + (size_t)grow * d + gcol;
#pragma unroll
        for (int r = 0; r < 8; ++r) {
          const float t = acc[tm][tn][r];
          s2p += t * t;
          s3p += (float)Sb[(size_t)r * d] * t;
        }
      }
    }
#pragma unroll
    for (int o = 16; o > 0; o >>= 1) {
      s2p += __shfl_xor(s2p, o, 32);
      s3p += __shfl_xor(s3p, o, 32);
    }
    if (lane == 0) {
      const int gw = blockIdx.x * 8 + wid;
      partA[gw] = s2p;
      partB[gw] = s3p;
    }
  } else {
#pragma unroll
    for (int tm = 0; tm < 2; ++tm) {
#pragma unroll
      for (int tn = 0; tn < 2; ++tn) {
        const int grow = bi * 64 + wr * 32 + tm * 16 + sel * 8;
        const int gcol = bj * 128 + wc * 32 + tn * 16 + m;
        float* Ob = Out + (size_t)grow * d + gcol;
        if (MODE == 2) {
          const float* Xb = Xold + (size_t)grow * d + gcol;
#pragma unroll
          for (int r = 0; r < 8; ++r) Ob[(size_t)r * d] = 2.0f * Xb[(size_t)r * d] - acc[tm][tn][r];
        } else {
#pragma unroll
          for (int r = 0; r < 8; ++r) Ob[(size_t)r * d] = acc[tm][tn][r];
        }
      }
    }
  }
}

// ---------------- final reductions over X = pinv(Py) ----------------
__global__ __launch_bounds__(TPB) void k_reduce_X(const float* __restrict__ X, const float* __restrict__ y,
                                                  const float* __restrict__ ye, int d,
                                                  float* __restrict__ p1, float* __restrict__ p2) {
  __shared__ float r1[TPB], r2[TPB];
  const size_t n = (size_t)d * d;
  float a = 0.f, b = 0.f;
  for (size_t i = (size_t)blockIdx.x * TPB + threadIdx.x; i < n; i += (size_t)gridDim.x * TPB) {
    const int j = (int)(i % (size_t)d);
    const float v = X[i];
    a += v;
    b += v * (y[j] - ye[j]);
  }
  r1[threadIdx.x] = a; r2[threadIdx.x] = b; __syncthreads();
  for (int s = TPB / 2; s > 0; s >>= 1) {
    if ((int)threadIdx.x < s) {
      r1[threadIdx.x] += r1[threadIdx.x + s];
      r2[threadIdx.x] += r2[threadIdx.x + s];
    }
    __syncthreads();
  }
  if (threadIdx.x == 0) { p1[blockIdx.x] = r1[0]; p2[blockIdx.x] = r2[0]; }
}

// ---------------- outputs ----------------
__global__ __launch_bounds__(TPB) void k_out_x(const float* __restrict__ xp, const float* __restrict__ s3p,
                                               const float* __restrict__ sXwp, float* __restrict__ out, int d) {
  const int i = blockIdx.x * TPB + threadIdx.x;
  if (i < d) out[i] = xp[i] + (*s3p) * (*sXwp);
}

__global__ __launch_bounds__(TPB) void k_out_P(const float* __restrict__ Q, const float* __restrict__ s1p,
                                               const float* __restrict__ s3p, const float* __restrict__ sXp,
                                               float* __restrict__ out, size_t n) {
  const size_t idx = (size_t)blockIdx.x * TPB + threadIdx.x;
  if (idx < n) {
    const float s3 = *s3p;
    out[idx] = Q[idx] + (*s1p - s3 * s3 * (*sXp));
  }
}

// =====================================================================
extern "C" void kernel_launch(void* const* d_in, const int* in_sizes, int n_in,
                              void* d_out, int out_size, void* d_ws, size_t ws_size,
                              hipStream_t stream) {
  const float* x  = (const float*)d_in[0];
  const float* y  = (const float*)d_in[1];
  const float* u  = (const float*)d_in[2];
  const float* P  = (const float*)d_in[3];
  const float* Q  = (const float*)d_in[4];
  const float* R  = (const float*)d_in[5];
  const float* A  = (const float*)d_in[6];
  const float* B  = (const float*)d_in[7];
  const float* C  = (const float*)d_in[8];
  const float* D  = (const float*)d_in[9];
  const float* c1 = (const float*)d_in[10];
  const float* c2 = (const float*)d_in[11];
  const int d = in_sizes[0];
  const size_t dd = (size_t)d * d;
  const float w = 1.0f / (2.0f * (float)d);

  // ---- workspace carve-up (256B aligned) ----
  char* wsb = (char*)d_ws;
  size_t off = 0;
  auto give = [&](size_t bytes) -> char* {
    char* p = wsb + off;
    off = (off + bytes + 255) & ~(size_t)255;
    return p;
  };
  float*    scal  = (float*)give(64 * sizeof(float));
  float*    xp    = (float*)give((size_t)d * 4);
  float*    yest  = (float*)give((size_t)d * 4);
  float*    rs    = (float*)give((size_t)d * 4);
  float*    cs    = (float*)give((size_t)d * 4);
  float*    part1 = (float*)give(4096 * 4);
  float*    part2 = (float*)give(4096 * 4);
  const int gblocks = (d / 64) * (d / 128);   // GEMM grid
  const int gparts  = gblocks * 8;            // one partial per wave
  float*    partA = (float*)give((size_t)gparts * 4);
  float*    partB = (float*)give((size_t)gparts * 4);
  _Float16* S2h   = (_Float16*)give(dd * 2);
  _Float16* Ch    = (_Float16*)give(dd * 2);
  _Float16* PyH   = (_Float16*)give(dd * 2);
  _Float16* Xh    = (_Float16*)give(dd * 2);
  _Float16* XTh   = (_Float16*)give(dd * 2);
  _Float16* ZTh   = (_Float16*)give(dd * 2);
  float*    Zf    = (float*)give(dd * 4);
  float*    X0    = (float*)give(dd * 4);
  float*    X1    = (float*)give(dd * 4);
  (void)ws_size; (void)n_in; (void)out_size;

  float* sumP  = scal + 0;
  float* s1    = scal + 1;
  float* s2    = scal + 2;
  float* s3    = scal + 3;
  float* alpha = scal + 4;
  float* sX    = scal + 5;
  float* sXw   = scal + 6;
  float* a2    = scal + 7;
  float* a3    = scal + 8;

  const int eblocks = (int)(dd / TPB);

  // 1) s1 = 2*w^2*sum(P)
  k_sum_partial<<<1024, TPB, 0, stream>>>(P, dd, part1);
  k_freduce<<<1, TPB, 0, stream>>>(part1, 1024, sumP);
  k_scalar_s1<<<1, 1, 0, stream>>>(sumP, w, s1);

  // 2) xp = A x + B u + c1  (== x_est)
  k_matvec2<<<d, TPB, 0, stream>>>(A, x, B, u, c1, xp, d);

  // 3) f16 operands for T = S2 * C^T
  k_build_s2h<<<eblocks, TPB, 0, stream>>>(Q, s1, w, S2h, d);
  k_f2h<<<eblocks, TPB, 0, stream>>>(C, Ch, dd);

  // 4) fused WMMA GEMM-reduction: s2 = 2w*sum(T^2), s3 = 2w*sum(S2.*T)
  k_gemm_tiled<0><<<gblocks, TPB, 0, stream>>>(S2h, Ch, d, S2h, partA, partB, nullptr, nullptr);
  k_freduce<<<1, TPB, 0, stream>>>(partA, gparts, a2);
  k_freduce<<<1, TPB, 0, stream>>>(partB, gparts, a3);
  k_scalar_s23<<<1, 1, 0, stream>>>(a2, a3, w, s2, s3);

  // 5) y_est = C x_est + D u + c2
  k_matvec2<<<d, TPB, 0, stream>>>(C, xp, D, u, c2, yest, d);

  // 6) Newton-Schulz init: X0 = Py^T / (||Py||_1 * ||Py||_inf), Py = R + s2
  k_rowsum<<<d, TPB, 0, stream>>>(R, rs, d);
  k_colsum<<<d, TPB, 0, stream>>>(R, cs, d);
  k_alpha<<<1, TPB, 0, stream>>>(rs, cs, s2, d, alpha);
  k_build_pyh<<<eblocks, TPB, 0, stream>>>(R, s2, PyH, dd);
  k_init_X<<<eblocks, TPB, 0, stream>>>(R, s2, alpha, X0, d);

  // 7) Newton-Schulz iterations: X <- 2X - X*(Py*X)   (WMMA f16->f32)
  const int ITER = 12;
  float* Xc = X0;
  float* Xn = X1;
  for (int it = 0; it < ITER; ++it) {
    k_conv_XT<<<eblocks, TPB, 0, stream>>>(Xc, Xh, XTh, d);
    k_gemm_tiled<1><<<gblocks, TPB, 0, stream>>>(PyH, XTh, d, nullptr, nullptr, nullptr, nullptr, Zf);
    k_convT<<<eblocks, TPB, 0, stream>>>(Zf, ZTh, d);
    k_gemm_tiled<2><<<gblocks, TPB, 0, stream>>>(Xh, ZTh, d, nullptr, nullptr, nullptr, Xc, Xn);
    float* t = Xc; Xc = Xn; Xn = t;
  }

  // 8) sum(X) and sum_{ij} X[i][j]*(y - y_est)_j
  k_reduce_X<<<2048, TPB, 0, stream>>>(Xc, y, yest, d, part1, part2);
  k_freduce<<<1, TPB, 0, stream>>>(part1, 2048, sX);
  k_freduce<<<1, TPB, 0, stream>>>(part2, 2048, sXw);

  // 9) outputs: x_new (d), P_new (d*d)
  float* out = (float*)d_out;
  k_out_x<<<(d + TPB - 1) / TPB, TPB, 0, stream>>>(xp, s3, sXw, out, d);
  k_out_P<<<eblocks, TPB, 0, stream>>>(Q, s1, s3, sX, out + d, dd);
}
